// DeepM3Model_27032524161052
// MI455X (gfx1250) — compile-verified, matrix operand order checked by reading, not run on hardware
//
#include <hip/hip_runtime.h>

typedef _Float16 half_t;
typedef __attribute__((ext_vector_type(16))) _Float16 v16h;
typedef __attribute__((ext_vector_type(8)))  float    v8f;

#define Bsz 512
#define Lsz 128
#define Dsz 512
#define LDA 528   // padded LDS row stride in halves (breaks bank conflicts, keeps 32B align)

__device__ __forceinline__ v8f wmma_f16(v16h a, v16h b, v8f c) {
  // D = A(16x32 f16) * B(32x16 f16) + C(16x16 f32)
  return __builtin_amdgcn_wmma_f32_16x16x32_f16(false, a, false, b, (short)0, c, false, false);
}

// ---------------------------------------------------------------------------
// Weight conversion: fp32 -> fp16 into workspace
// ---------------------------------------------------------------------------
__global__ __launch_bounds__(256) void convert_weights_kernel(
    const float* __restrict__ W1, const float* __restrict__ W2,
    const float* __restrict__ Wih, const float* __restrict__ Whh,
    half_t* __restrict__ o1, half_t* __restrict__ o2,
    half_t* __restrict__ oih, half_t* __restrict__ ohh)
{
  const int n1 = Dsz * Dsz;       // 262144
  const int n2 = 3 * Dsz * Dsz;   // 786432
  int idx = blockIdx.x * 256 + threadIdx.x;
  if (idx < n1)                 o1[idx]              = (half_t)W1[idx];
  else if (idx < 2 * n1)        o2[idx - n1]         = (half_t)W2[idx - n1];
  else if (idx < 2 * n1 + n2)   oih[idx - 2 * n1]    = (half_t)Wih[idx - 2 * n1];
  else if (idx < 2 * n1 + 2*n2) ohh[idx - 2*n1 - n2] = (half_t)Whh[idx - 2*n1 - n2];
}

// ---------------------------------------------------------------------------
// 16xK * KxN GEMM fragment: wave computes C[16 x 64] at columns nb..nb+63.
// A from LDS (stride LDA), B = weight[out][in] row-major in global (f16).
// ---------------------------------------------------------------------------
__device__ __forceinline__ void gemm4(const half_t* __restrict__ A,
                                      const half_t* __restrict__ W,
                                      int nb, int r, int hi, v8f acc[4])
{
  const half_t* arow  = A + r * LDA + hi * 16;
  const half_t* wbase = W + (size_t)(nb + r) * Dsz + hi * 16;
#pragma unroll 4
  for (int kk = 0; kk < Dsz / 32; ++kk) {
    v16h a = *(const v16h*)(arow + kk * 32);
#pragma unroll
    for (int nt = 0; nt < 4; ++nt) {
      v16h b = *(const v16h*)(wbase + (size_t)nt * 16 * Dsz + kk * 32);
      acc[nt] = wmma_f16(a, b, acc[nt]);
    }
  }
}

// Fused GRU GEMM chunk: computes gi (A=X) and gh (A=H) for 3 gates x 2 n-tiles.
__device__ __forceinline__ void gemm_gru(const half_t* __restrict__ X,
                                         const half_t* __restrict__ H,
                                         const half_t* __restrict__ Wih,
                                         const half_t* __restrict__ Whh,
                                         int nb, int ch, int r, int hi,
                                         v8f gia[3][2], v8f gha[3][2])
{
  const half_t* xr = X + r * LDA + hi * 16;
  const half_t* hr = H + r * LDA + hi * 16;
#pragma unroll 2
  for (int kk = 0; kk < Dsz / 32; ++kk) {
    v16h ax = *(const v16h*)(xr + kk * 32);
    v16h ah = *(const v16h*)(hr + kk * 32);
#pragma unroll
    for (int g = 0; g < 3; ++g) {
#pragma unroll
      for (int tt = 0; tt < 2; ++tt) {
        int nrow = g * Dsz + nb + (ch * 2 + tt) * 16 + r;
        v16h bi = *(const v16h*)(Wih + (size_t)nrow * Dsz + hi * 16 + kk * 32);
        gia[g][tt] = wmma_f16(ax, bi, gia[g][tt]);
        v16h bh = *(const v16h*)(Whh + (size_t)nrow * Dsz + hi * 16 + kk * 32);
        gha[g][tt] = wmma_f16(ah, bh, gha[g][tt]);
      }
    }
  }
}

__device__ __forceinline__ float sigm(float v) { return 1.f / (1.f + __expf(-v)); }

// ---------------------------------------------------------------------------
// Main kernel: one block = 16 batch rows, full 128-step scan.
// 256 threads = 8 waves; wave w owns hidden columns [64w, 64w+64).
// ---------------------------------------------------------------------------
__global__ __launch_bounds__(256, 1) void node_gru_scan_kernel(
    const float* __restrict__ tarr, const float* __restrict__ emb,
    const float* __restrict__ b_ih, const float* __restrict__ b_hh,
    const float* __restrict__ b1,   const float* __restrict__ b2,
    const int*   __restrict__ x,
    const half_t* __restrict__ W1h, const half_t* __restrict__ W2h,
    const half_t* __restrict__ Wihh, const half_t* __restrict__ Whhh,
    float* __restrict__ out)
{
  __shared__ __align__(32) half_t a16[16 * LDA];  // GEMM-A operand (y / h)
  __shared__ __align__(32) half_t u16[16 * LDA];  // tanh intermediate
  __shared__ __align__(32) half_t x16[16 * LDA];  // embedded event (f16)
  __shared__ float dtv[16];
  __shared__ float mv[16];

  const int tid   = threadIdx.x;
  const int lane  = tid & 31;
  const int wave  = tid >> 5;
  const int r     = lane & 15;   // A-row / B-col within tile
  const int hi    = lane >> 4;   // K-half selector
  const int nb    = wave * 64;   // wave's column base
  const int rbase = blockIdx.x * 16;

  const v8f vz = {0.f, 0.f, 0.f, 0.f, 0.f, 0.f, 0.f, 0.f};
  v8f hreg[4];                   // master h (f32), C-fragment layout
#pragma unroll
  for (int nt = 0; nt < 4; ++nt) hreg[nt] = vz;

  for (int idx = tid; idx < 16 * LDA; idx += 256) a16[idx] = (half_t)0.f;
  __syncthreads();

  for (int step = 0; step < Lsz; ++step) {
    // ---- P0: stage x embedding (f32 -> f16), per-row dt / mask ----
    {
      int rr = tid >> 4, c0 = (tid & 15) * 32;
      int xi = x[(rbase + rr) * Lsz + step];
      const float* er = emb + (size_t)xi * Dsz + c0;
#pragma unroll 8
      for (int j = 0; j < 32; ++j) x16[rr * LDA + c0 + j] = (half_t)er[j];
      if (tid < 16) {
        int xj = x[(rbase + tid) * Lsz + step];
        mv[tid] = (xj != 0) ? 1.f : 0.f;
        float dtt = 0.f;
        if (step > 0)
          dtt = tarr[(rbase + tid) * Lsz + step] - tarr[(rbase + tid) * Lsz + step - 1];
        dtv[tid] = fmaxf(dtt, 1e-6f);
      }
    }
    __syncthreads();

    float dloc[8], mloc[8];
#pragma unroll
    for (int j = 0; j < 8; ++j) { dloc[j] = dtv[hi * 8 + j]; mloc[j] = mv[hi * 8 + j]; }
    const float gateev = (step > 0) ? 1.f : 0.f;

    // ---- RK4: h_ev = h + dt/6 * (k1 + 2k2 + 2k3 + k4), f = tanh(.W1^T+b1).W2^T+b2 ----
    v8f ks[4];
#pragma unroll
    for (int nt = 0; nt < 4; ++nt) ks[nt] = vz;

#pragma unroll
    for (int s = 1; s <= 4; ++s) {
      // GEMM1: y @ W1^T  -> tanh -> u16
      v8f acc[4];
#pragma unroll
      for (int nt = 0; nt < 4; ++nt) acc[nt] = vz;
      gemm4(a16, W1h, nb, r, hi, acc);
#pragma unroll
      for (int nt = 0; nt < 4; ++nt) {
        int n = nb + nt * 16 + r;
        float b1v = b1[n];
#pragma unroll
        for (int j = 0; j < 8; ++j)
          u16[(hi * 8 + j) * LDA + n] = (half_t)tanhf(acc[nt][j] + b1v);
      }
      __syncthreads();

      // GEMM2: u @ W2^T -> k_s; update ksum; write next y (or blended h) to a16
      v8f acc2[4];
#pragma unroll
      for (int nt = 0; nt < 4; ++nt) acc2[nt] = vz;
      gemm4(u16, W2h, nb, r, hi, acc2);
#pragma unroll
      for (int nt = 0; nt < 4; ++nt) {
        int n = nb + nt * 16 + r;
        float b2v = b2[n];
#pragma unroll
        for (int j = 0; j < 8; ++j) {
          int m = hi * 8 + j;
          float k = acc2[nt][j] + b2v;
          if (s == 1)      ks[nt][j]  = k;
          else if (s == 4) ks[nt][j] += k;
          else             ks[nt][j] += 2.f * k;
          if (s < 4) {
            float cs = (s == 3) ? 1.f : 0.5f;
            float y  = hreg[nt][j] + cs * dloc[j] * k;
            a16[m * LDA + n] = (half_t)y;
          } else {
            float hev = hreg[nt][j] + (dloc[j] * (1.f / 6.f)) * ks[nt][j];
            float em  = gateev * mloc[j];
            float hu  = em * hev + (1.f - em) * hreg[nt][j];
            hreg[nt][j] = hu;               // h_used
            a16[m * LDA + n] = (half_t)hu;  // A operand for gh GEMM
          }
        }
      }
      __syncthreads();
    }

    // ---- GRUCell: gi = x@Wih^T+b_ih, gh = h@Whh^T+b_hh, gates r,z,n ----
#pragma unroll 1
    for (int ch = 0; ch < 2; ++ch) {
      v8f gia[3][2], gha[3][2];
#pragma unroll
      for (int g = 0; g < 3; ++g)
#pragma unroll
        for (int tt = 0; tt < 2; ++tt) { gia[g][tt] = vz; gha[g][tt] = vz; }
      gemm_gru(x16, a16, Wihh, Whhh, nb, ch, r, hi, gia, gha);
#pragma unroll
      for (int tt = 0; tt < 2; ++tt) {
        int nt = ch * 2 + tt;
        int n  = nb + nt * 16 + r;
        float bir = b_ih[n], biz = b_ih[Dsz + n], bin_ = b_ih[2 * Dsz + n];
        float bhr = b_hh[n], bhz = b_hh[Dsz + n], bhn_ = b_hh[2 * Dsz + n];
#pragma unroll
        for (int j = 0; j < 8; ++j) {
          float ir  = gia[0][tt][j] + bir;
          float iz  = gia[1][tt][j] + biz;
          float in_ = gia[2][tt][j] + bin_;
          float hr_ = gha[0][tt][j] + bhr;
          float hz  = gha[1][tt][j] + bhz;
          float hn  = gha[2][tt][j] + bhn_;
          float rg  = sigm(ir + hr_);
          float zg  = sigm(iz + hz);
          float ng  = tanhf(in_ + rg * hn);
          float hu  = hreg[nt][j];
          float hnew = (1.f - zg) * ng + zg * hu;
          hreg[nt][j] = mloc[j] * hnew + (1.f - mloc[j]) * hu;
        }
      }
    }

    __syncthreads();  // all waves done reading a16 (old h) before overwrite
#pragma unroll
    for (int nt = 0; nt < 4; ++nt) {
      int n = nb + nt * 16 + r;
#pragma unroll
      for (int j = 0; j < 8; ++j)
        a16[(hi * 8 + j) * LDA + n] = (half_t)hreg[nt][j];
    }
    __syncthreads();
  }

  // ---- output: h [B, D] fp32 ----
#pragma unroll
  for (int nt = 0; nt < 4; ++nt) {
    int n = nb + nt * 16 + r;
#pragma unroll
    for (int j = 0; j < 8; ++j)
      out[(size_t)(rbase + hi * 8 + j) * Dsz + n] = hreg[nt][j];
  }
}

// ---------------------------------------------------------------------------
extern "C" void kernel_launch(void* const* d_in, const int* in_sizes, int n_in,
                              void* d_out, int out_size, void* d_ws, size_t ws_size,
                              hipStream_t stream) {
  (void)in_sizes; (void)n_in; (void)out_size; (void)ws_size;
  // setup_inputs order: t, emb, W_ih, W_hh, b_ih, b_hh, W1, b1, W2, b2, x
  const float* t    = (const float*)d_in[0];
  const float* emb  = (const float*)d_in[1];
  const float* W_ih = (const float*)d_in[2];
  const float* W_hh = (const float*)d_in[3];
  const float* b_ih = (const float*)d_in[4];
  const float* b_hh = (const float*)d_in[5];
  const float* W1   = (const float*)d_in[6];
  const float* b1   = (const float*)d_in[7];
  const float* W2   = (const float*)d_in[8];
  const float* b2   = (const float*)d_in[9];
  const int*   x    = (const int*)d_in[10];
  float* out = (float*)d_out;

  // workspace layout: f16 weights (4.2 MB total)
  half_t* W1h  = (half_t*)d_ws;
  half_t* W2h  = W1h + (size_t)Dsz * Dsz;
  half_t* Wihh = W2h + (size_t)Dsz * Dsz;
  half_t* Whhh = Wihh + (size_t)3 * Dsz * Dsz;

  const int total = 2 * Dsz * Dsz + 2 * 3 * Dsz * Dsz;  // 2097152
  convert_weights_kernel<<<(total + 255) / 256, 256, 0, stream>>>(
      W1, W2, W_ih, W_hh, W1h, W2h, Wihh, Whhh);

  node_gru_scan_kernel<<<Bsz / 16, 256, 0, stream>>>(
      t, emb, b_ih, b_hh, b1, b2, x, W1h, W2h, Wihh, Whhh, out);
}
